// EntityLinking_24713241821559
// MI455X (gfx1250) — matrix operand order repaired and checked by reading, not compiled, not run on hardware
//
#include <hip/hip_runtime.h>

// ---------------------------------------------------------------------------
// Bilinear entity-linking scorer for MI455X (gfx1250, wave32).
// scores = (emb_a @ W) @ emb_b^T + b ; row-wise max/argmax/valid.
// Strategy: fp32 -> bf16 hi/lo split, 3x V_WMMA_F32_16X16X32_BF16 per K=32
// slab (hi*hi + hi*lo + lo*hi) for ~fp32 accuracy at bf16-WMMA throughput.
// GEMM2 is register-blocked (2 row-strips x 4 j-tiles = 8 accumulators per
// wave) with the shared A-strip staged in LDS to cut L2 traffic ~32x.
// ---------------------------------------------------------------------------

#define N_ROWS 4096
#define M_COLS 4096
#define H_DIM  768
#define JCHUNKS 8

typedef __attribute__((ext_vector_type(16))) __bf16 v16bf;
typedef __attribute__((ext_vector_type(8)))  float  v8f;

using u16 = unsigned short;

union FragU {
    uint4  u[2];
    v16bf  v;
};

__device__ __forceinline__ u16 f2bf(float f) {
    unsigned int u = __float_as_uint(f);
    // round-to-nearest-even bf16
    unsigned int r = (u + 0x7FFFu + ((u >> 16) & 1u)) >> 16;
    return (u16)r;
}
__device__ __forceinline__ float bf2f(u16 h) {
    return __uint_as_float(((unsigned int)h) << 16);
}

// A operand (16x32 bf16, MxK): lane holds row M=lane&15.
// lanes 0-15: K = k..k+7 (VGPR0-3) and k+16..k+23 (VGPR4-7)
// lanes 16-31: K = k+8..k+15 and k+24..k+31
__device__ __forceinline__ v16bf load_fragA(const u16* rowptr, int k, int lane) {
    int kb = k + ((lane & 16) ? 8 : 0);
    FragU f;
    f.u[0] = *(const uint4*)(rowptr + kb);
    f.u[1] = *(const uint4*)(rowptr + kb + 16);
    return f.v;
}

// B operand (32x16 bf16, KxN): lane holds column N=lane&15 (a row of the
// transposed buffer). lanes 0-15: K=k..k+7 and k+16..k+23; lanes 16-31:
// K=k+8..k+15 and k+24..k+31.
__device__ __forceinline__ v16bf load_fragB(const u16* rowptr, int k, int lane) {
    int kb = k + ((lane & 16) ? 8 : 0);
    FragU f;
    f.u[0] = *(const uint4*)(rowptr + kb);
    f.u[1] = *(const uint4*)(rowptr + kb + 16);
    return f.v;
}

__device__ __forceinline__ v8f wmma_bf16(v16bf a, v16bf b, v8f c) {
    return __builtin_amdgcn_wmma_f32_16x16x32_bf16(
        /*neg_a=*/false, a, /*neg_b=*/false, b,
        /*c_mod=*/(short)0, c, /*reuse_a=*/false, /*reuse_b=*/false);
}

// ---------------------------------------------------------------------------
// Kernel 1: split fp32 inputs into bf16 hi/lo. W is transposed on the fly so
// GEMM1 B-operand columns become contiguous rows.
// ---------------------------------------------------------------------------
__global__ void k_split_bf16(const float* __restrict__ A,   // emb_a [N,H]
                             const float* __restrict__ B,   // emb_b [M,H]
                             const float* __restrict__ W,   // W [H,H]
                             u16* __restrict__ a_hi, u16* __restrict__ a_lo,
                             u16* __restrict__ b_hi, u16* __restrict__ b_lo,
                             u16* __restrict__ wt_hi, u16* __restrict__ wt_lo) {
    const int NH = N_ROWS * H_DIM;
    const int MH = M_COLS * H_DIM;
    const int HH = H_DIM * H_DIM;
    int i = blockIdx.x * blockDim.x + threadIdx.x;
    if (i < NH) {
        float v = A[i];
        u16 h = f2bf(v);
        a_hi[i] = h;
        a_lo[i] = f2bf(v - bf2f(h));
    } else if (i < NH + MH) {
        int j = i - NH;
        float v = B[j];
        u16 h = f2bf(v);
        b_hi[j] = h;
        b_lo[j] = f2bf(v - bf2f(h));
    } else if (i < NH + MH + HH) {
        int j = i - NH - MH;
        int k = j / H_DIM;         // row of W
        int n = j % H_DIM;         // col of W
        float v = W[k * H_DIM + n];
        u16 h = f2bf(v);
        wt_hi[n * H_DIM + k] = h;  // store W^T
        wt_lo[n * H_DIM + k] = f2bf(v - bf2f(h));
    }
}

// ---------------------------------------------------------------------------
// Kernel 2: P = emb_a @ W  (fp32 via 3-term bf16 WMMA), output re-split to
// bf16 hi/lo for GEMM2. Each wave owns one 16x16 tile of P.
// grid = (N/16, H/128), block = 256 (8 waves).
// ---------------------------------------------------------------------------
__global__ void k_gemm1_wmma(const u16* __restrict__ a_hi, const u16* __restrict__ a_lo,
                             const u16* __restrict__ wt_hi, const u16* __restrict__ wt_lo,
                             u16* __restrict__ p_hi, u16* __restrict__ p_lo) {
    const int lane = threadIdx.x & 31;
    const int wave = threadIdx.x >> 5;
    const int row0 = blockIdx.x * 16;
    const int col0 = blockIdx.y * 128 + wave * 16;

    const u16* arh = a_hi + (size_t)(row0 + (lane & 15)) * H_DIM;
    const u16* arl = a_lo + (size_t)(row0 + (lane & 15)) * H_DIM;
    const u16* brh = wt_hi + (size_t)(col0 + (lane & 15)) * H_DIM;
    const u16* brl = wt_lo + (size_t)(col0 + (lane & 15)) * H_DIM;

    v8f acc = {};
#pragma unroll 4
    for (int k = 0; k < H_DIM; k += 32) {
        v16bf ah = load_fragA(arh, k, lane);
        v16bf al = load_fragA(arl, k, lane);
        v16bf bh = load_fragB(brh, k, lane);
        v16bf bl = load_fragB(brl, k, lane);
        acc = wmma_bf16(ah, bh, acc);
        acc = wmma_bf16(ah, bl, acc);
        acc = wmma_bf16(al, bh, acc);
    }

    const int half = (lane >> 4) & 1;
    const int c = col0 + (lane & 15);
#pragma unroll 8
    for (int i = 0; i < 8; ++i) {
        int r = row0 + i + half * 8;
        float v = acc[i];
        u16 h = f2bf(v);
        p_hi[(size_t)r * H_DIM + c] = h;
        p_lo[(size_t)r * H_DIM + c] = f2bf(v - bf2f(h));
    }
}

// ---------------------------------------------------------------------------
// Kernel 3: S-tile = P @ emb_b^T fused with running max/argmax.
// grid = (N/32 row tiles, JCHUNKS), block = 256 (8 waves).
// Block stages 32 rows of P (hi+lo, 96 KB) in LDS. Each wave computes
// 2 row-strips x 4 j-blocks = 8 accumulators: one A-fragment pair feeds 12
// WMMAs, one B-fragment pair feeds 6 WMMAs. LDS is re-used (post-barrier)
// for the cross-wave max/argmax reduction.
// ---------------------------------------------------------------------------
__global__ void __launch_bounds__(256)
k_gemm2_argmax_wmma(const u16* __restrict__ p_hi, const u16* __restrict__ p_lo,
                    const u16* __restrict__ b_hi, const u16* __restrict__ b_lo,
                    float* __restrict__ part_best, int* __restrict__ part_idx) {
    // 2 * 32 rows * 768 u16 = 96 KB (hi strip then lo strip)
    __shared__ u16 s_a[2 * 32 * H_DIM];

    const int tid  = threadIdx.x;
    const int lane = tid & 31;
    const int wave = tid >> 5;
    const int row0 = blockIdx.x * 32;
    const int chunk = blockIdx.y;
    const int jBase0 = chunk * (M_COLS / JCHUNKS);   // 512 columns per chunk

    // ---- stage 32 rows of p_hi / p_lo into LDS (uint4 = 8 u16 per copy) ----
    {
        const int VEC_PER_ROW = H_DIM / 8;           // 96
        const int TOTAL = 32 * VEC_PER_ROW;          // 3072 uint4 per buffer
        uint4* s_hi4 = (uint4*)s_a;
        uint4* s_lo4 = (uint4*)(s_a + 32 * H_DIM);
        for (int e = tid; e < TOTAL; e += 256) {
            int r  = e / VEC_PER_ROW;
            int cu = e - r * VEC_PER_ROW;
            s_hi4[e] = *(const uint4*)(p_hi + (size_t)(row0 + r) * H_DIM + cu * 8);
            s_lo4[e] = *(const uint4*)(p_lo + (size_t)(row0 + r) * H_DIM + cu * 8);
        }
    }
    __syncthreads();

    // per-lane A row pointers in LDS for the two row-strips
    const u16* arh0 = s_a + (size_t)(0 * 16 + (lane & 15)) * H_DIM;
    const u16* arh1 = s_a + (size_t)(1 * 16 + (lane & 15)) * H_DIM;
    const u16* arl0 = s_a + 32 * H_DIM + (size_t)(0 * 16 + (lane & 15)) * H_DIM;
    const u16* arl1 = s_a + 32 * H_DIM + (size_t)(1 * 16 + (lane & 15)) * H_DIM;

    // B row pointers for this wave's 4 j-blocks
    const u16* brhp[4];
    const u16* brlp[4];
    int jb[4];
#pragma unroll
    for (int t = 0; t < 4; ++t) {
        jb[t] = jBase0 + (wave + 8 * t) * 16;
        brhp[t] = b_hi + (size_t)(jb[t] + (lane & 15)) * H_DIM;
        brlp[t] = b_lo + (size_t)(jb[t] + (lane & 15)) * H_DIM;
    }

    v8f acc[2][4];
#pragma unroll
    for (int s = 0; s < 2; ++s)
#pragma unroll
        for (int t = 0; t < 4; ++t) acc[s][t] = (v8f){};

#pragma unroll 1
    for (int k = 0; k < H_DIM; k += 32) {
        v16bf ah0 = load_fragA(arh0, k, lane);
        v16bf al0 = load_fragA(arl0, k, lane);
        v16bf ah1 = load_fragA(arh1, k, lane);
        v16bf al1 = load_fragA(arl1, k, lane);
#pragma unroll
        for (int t = 0; t < 4; ++t) {
            v16bf bh = load_fragB(brhp[t], k, lane);
            v16bf bl = load_fragB(brlp[t], k, lane);
            acc[0][t] = wmma_bf16(ah0, bh, acc[0][t]);
            acc[0][t] = wmma_bf16(ah0, bl, acc[0][t]);
            acc[0][t] = wmma_bf16(al0, bh, acc[0][t]);
            acc[1][t] = wmma_bf16(ah1, bh, acc[1][t]);
            acc[1][t] = wmma_bf16(ah1, bl, acc[1][t]);
            acc[1][t] = wmma_bf16(al1, bh, acc[1][t]);
        }
    }

    // ---- per-lane running max/argmax over this wave's 4 j-blocks ----
    float best[2][8];
    int   bidx[2][8];
#pragma unroll
    for (int s = 0; s < 2; ++s)
#pragma unroll
        for (int i = 0; i < 8; ++i) { best[s][i] = -__builtin_inff(); bidx[s][i] = 0; }

#pragma unroll
    for (int t = 0; t < 4; ++t) {
        const int col = jb[t] + (lane & 15);
#pragma unroll
        for (int s = 0; s < 2; ++s)
#pragma unroll
            for (int i = 0; i < 8; ++i) {
                float v = acc[s][t][i];
                if (v > best[s][i]) { best[s][i] = v; bidx[s][i] = col; }
            }
    }

    // ---- cross-lane / cross-wave reduction; alias LDS staging buffer ----
    __syncthreads();   // all waves done reading staged A
    float* s_bestF = (float*)s_a;                       // [8][32][16] = 16 KB
    int*   s_idxI  = (int*)((char*)s_a + 8 * 32 * 16 * 4);  // next 16 KB

    const int half = (lane >> 4) & 1;
    const int cl = lane & 15;
#pragma unroll
    for (int s = 0; s < 2; ++s)
#pragma unroll
        for (int i = 0; i < 8; ++i) {
            int r = s * 16 + i + half * 8;              // 0..31
            s_bestF[(wave * 32 + r) * 16 + cl] = best[s][i];
            s_idxI [(wave * 32 + r) * 16 + cl] = bidx[s][i];
        }
    __syncthreads();

    if (tid < 32) {
        const int r = tid;
        float bb = -__builtin_inff();
        int   bi = 0x7FFFFFFF;
        for (int w = 0; w < 8; ++w) {
            for (int c = 0; c < 16; ++c) {
                float v = s_bestF[(w * 32 + r) * 16 + c];
                int  ix = s_idxI [(w * 32 + r) * 16 + c];
                if (v > bb || (v == bb && ix < bi)) { bb = v; bi = ix; }
            }
        }
        const int grow = row0 + r;
        part_best[grow * JCHUNKS + chunk] = bb;
        part_idx [grow * JCHUNKS + chunk] = bi;
    }
}

// ---------------------------------------------------------------------------
// Kernel 4: combine partials across j-chunks, add bias, emit outputs:
// d_out[0:N) = best_scores, [N:2N) = best_idx (as float), [2N:3N) = valid.
// ---------------------------------------------------------------------------
__global__ void k_combine(const float* __restrict__ part_best, const int* __restrict__ part_idx,
                          const float* __restrict__ bias, float* __restrict__ out) {
    int r = blockIdx.x * blockDim.x + threadIdx.x;
    if (r >= N_ROWS) return;
    float bb = -__builtin_inff();
    int   bi = 0x7FFFFFFF;
#pragma unroll
    for (int c = 0; c < JCHUNKS; ++c) {
        float v = part_best[r * JCHUNKS + c];
        int  ix = part_idx[r * JCHUNKS + c];
        if (v > bb || (v == bb && ix < bi)) { bb = v; bi = ix; }
    }
    float s = bb + bias[0];
    out[r]              = s;
    out[N_ROWS + r]     = (float)bi;
    out[2 * N_ROWS + r] = (s > 0.0f) ? 1.0f : 0.0f;
}

// ---------------------------------------------------------------------------
extern "C" void kernel_launch(void* const* d_in, const int* in_sizes, int n_in,
                              void* d_out, int out_size, void* d_ws, size_t ws_size,
                              hipStream_t stream) {
    const float* emb_a = (const float*)d_in[0];   // [N, H]
    const float* emb_b = (const float*)d_in[1];   // [M, H]
    const float* W     = (const float*)d_in[2];   // [H, H]
    const float* bias  = (const float*)d_in[3];   // [1]
    float* out = (float*)d_out;

    const size_t NH = (size_t)N_ROWS * H_DIM;
    const size_t MH = (size_t)M_COLS * H_DIM;
    const size_t HH = (size_t)H_DIM * H_DIM;

    char* ws = (char*)d_ws;
    size_t off = 0;
    u16* a_hi  = (u16*)(ws + off); off += NH * sizeof(u16);
    u16* a_lo  = (u16*)(ws + off); off += NH * sizeof(u16);
    u16* b_hi  = (u16*)(ws + off); off += MH * sizeof(u16);
    u16* b_lo  = (u16*)(ws + off); off += MH * sizeof(u16);
    u16* wt_hi = (u16*)(ws + off); off += HH * sizeof(u16);
    u16* wt_lo = (u16*)(ws + off); off += HH * sizeof(u16);
    u16* p_hi  = (u16*)(ws + off); off += NH * sizeof(u16);
    u16* p_lo  = (u16*)(ws + off); off += NH * sizeof(u16);
    float* part_best = (float*)(ws + off); off += (size_t)N_ROWS * JCHUNKS * sizeof(float);
    int*   part_idx  = (int*)  (ws + off); off += (size_t)N_ROWS * JCHUNKS * sizeof(int);

    // 1) fp32 -> bf16 hi/lo split (+ W transpose)
    {
        size_t total = NH + MH + HH;
        int blocks = (int)((total + 255) / 256);
        k_split_bf16<<<blocks, 256, 0, stream>>>(emb_a, emb_b, W,
                                                 a_hi, a_lo, b_hi, b_lo, wt_hi, wt_lo);
    }
    // 2) P = emb_a @ W (WMMA), split to hi/lo
    {
        dim3 grid(N_ROWS / 16, H_DIM / 128);
        k_gemm1_wmma<<<grid, 256, 0, stream>>>(a_hi, a_lo, wt_hi, wt_lo, p_hi, p_lo);
    }
    // 3) S = P @ emb_b^T (WMMA, LDS-staged A, 8 accumulators/wave) + argmax
    {
        dim3 grid(N_ROWS / 32, JCHUNKS);
        k_gemm2_argmax_wmma<<<grid, 256, 0, stream>>>(p_hi, p_lo, b_hi, b_lo,
                                                      part_best, part_idx);
    }
    // 4) combine partials, add bias, write outputs
    {
        k_combine<<<(N_ROWS + 255) / 256, 256, 0, stream>>>(part_best, part_idx, bias, out);
    }
}
